// MaskedMultiHeadAttention_8778913153478
// MI455X (gfx1250) — matrix operand order repaired and checked by reading, not compiled, not run on hardware
//
#include <hip/hip_runtime.h>
#include <hip/hip_bf16.h>
#include <cstdint>

#define D_MODEL 1024
#define N_HEADS 16
#define HEAD_DIM 64
#define SEQ 2048
#define BATCH 2
#define M_TOTAL (BATCH * SEQ)   // 4096

typedef __attribute__((ext_vector_type(16))) __bf16 v16bf;
typedef __attribute__((ext_vector_type(8)))  float  v8f;

union FragBF {
    v16bf v;
    uint4 q[2];   // 32 bytes = 16 bf16
};

// ---------------------------------------------------------------------------
// CDNA5 async global->LDS copy (16B per lane), tracked with ASYNCcnt.
// GVS mode: global address = SGPR base + VGPR byte offset; LDS address taken from the vdst VGPR.
// ---------------------------------------------------------------------------
__device__ __forceinline__ void async_copy_b128(unsigned lds_byte_addr,
                                                const void* sbase,
                                                unsigned voff_bytes)
{
    asm volatile("global_load_async_to_lds_b128 %0, %1, %2 offset:0"
                 :: "v"(lds_byte_addr), "v"(voff_bytes), "s"(sbase)
                 : "memory");
}
__device__ __forceinline__ void wait_async()
{
    asm volatile("s_wait_asynccnt 0x0" ::: "memory");
}
__device__ __forceinline__ unsigned lds_addr_of(const void* p)
{
    return (unsigned)(uintptr_t)p;   // flat LDS addr: HW uses addr[31:0]
}

// ---------------------------------------------------------------------------
// fp32 -> bf16 elementwise preconversion
// ---------------------------------------------------------------------------
__global__ __launch_bounds__(256)
void to_bf16(const float* __restrict__ in, __bf16* __restrict__ out, int n)
{
    const int i = (blockIdx.x * 256 + threadIdx.x) * 4;
    if (i >= n) return;
    const float4 v = *(const float4*)(in + i);
    __bf16* o = out + i;
    o[0] = (__bf16)v.x; o[1] = (__bf16)v.y; o[2] = (__bf16)v.z; o[3] = (__bf16)v.w;
}

// ---------------------------------------------------------------------------
// GEMM: C[M,N] = A[M,K] * W[N,K]^T + bias[N]    (A,W bf16; C fp32)
// 128x128 block, K-step 32, 8 waves (4M x 2N), each wave 32x64 (2x4 WMMA tiles).
// Double-buffered LDS, async global->LDS staging, one barrier per K-step.
// ---------------------------------------------------------------------------
#define TB_M 128
#define TB_N 128
#define TB_K 32
#define LDSA 40   // padded row stride (bf16 elems)

__global__ __launch_bounds__(256)
void gemm_nt_bias(const __bf16* __restrict__ A,
                  const __bf16* __restrict__ W,
                  const float* __restrict__ bias,
                  float* __restrict__ C,
                  int M, int N, int K)
{
    __shared__ __bf16 As[2][TB_M * LDSA];
    __shared__ __bf16 Bs[2][TB_N * LDSA];

    const int tid  = threadIdx.x;
    const int lane = tid & 31;
    const int wave = tid >> 5;
    const int block_m = blockIdx.y * TB_M;
    const int block_n = blockIdx.x * TB_N;
    const int wm = (wave >> 1) * 32;
    const int wn = (wave & 1) * 64;

    v8f acc[2][4];
#pragma unroll
    for (int i = 0; i < 2; ++i)
#pragma unroll
        for (int j = 0; j < 4; ++j)
#pragma unroll
            for (int r = 0; r < 8; ++r) acc[i][j][r] = 0.0f;

    // staging task map: 512 chunks of 8 bf16 per tile, 2 per thread
    const int srow0 = tid >> 2;        // 0..63
    const int sq8   = (tid & 3) * 8;   // 0,8,16,24

    auto stage = [&](int buf, int kt) {
#pragma unroll
        for (int i = 0; i < 2; ++i) {
            const int row = srow0 + i * 64;
            const unsigned voffA = (unsigned)(((block_m + row) * K + kt + sq8) * 2);
            async_copy_b128(lds_addr_of(&As[buf][row * LDSA + sq8]), A, voffA);
            const unsigned voffW = (unsigned)(((block_n + row) * K + kt + sq8) * 2);
            async_copy_b128(lds_addr_of(&Bs[buf][row * LDSA + sq8]), W, voffW);
        }
    };

    const int nk = K / TB_K;
    stage(0, 0);

    const int mrow = lane & 15;
    const int k0a  = (lane < 16) ? 0 : 8;
    const int k0b  = (lane < 16) ? 0 : 8;

    for (int it = 0; it < nk; ++it) {
        wait_async();
        __syncthreads();
        if (it + 1 < nk) stage((it + 1) & 1, (it + 1) * TB_K);

        const __bf16* Ab = As[it & 1];
        const __bf16* Bb = Bs[it & 1];
        FragBF a[2], b[4];
#pragma unroll
        for (int i = 0; i < 2; ++i) {
            const __bf16* p = &Ab[(wm + i * 16 + mrow) * LDSA];
            a[i].q[0] = *(const uint4*)(p + k0a);
            a[i].q[1] = *(const uint4*)(p + k0a + 16);
        }
#pragma unroll
        for (int j = 0; j < 4; ++j) {
            const __bf16* p = &Bb[(wn + j * 16 + mrow) * LDSA + k0b];
            b[j].q[0] = *(const uint4*)(p);
            b[j].q[1] = *(const uint4*)(p + 16);
        }
#pragma unroll
        for (int i = 0; i < 2; ++i)
#pragma unroll
            for (int j = 0; j < 4; ++j)
                acc[i][j] = __builtin_amdgcn_wmma_f32_16x16x32_bf16(
                    false, a[i].v, false, b[j].v, (short)0, acc[i][j], false, false);
    }

    const int nl = lane & 15;
    const int mb = (lane >> 4) * 8;
#pragma unroll
    for (int i = 0; i < 2; ++i)
#pragma unroll
        for (int j = 0; j < 4; ++j) {
            const int ncol = block_n + wn + j * 16 + nl;
            const float bv = bias[ncol];
#pragma unroll
            for (int r = 0; r < 8; ++r) {
                const int mrw = block_m + wm + i * 16 + mb + r;
                C[(size_t)mrw * N + ncol] = acc[i][j][r] + bv;
            }
        }
}

// ---------------------------------------------------------------------------
// RoPE (reference's concat-style variant) + head-major bf16 repack.
// mode 0: Q (rope, *1/8),  mode 1: K (rope),  mode 2: V (plain)
// ---------------------------------------------------------------------------
__global__ __launch_bounds__(256)
void rope_pack(const float* __restrict__ P, __bf16* __restrict__ out, int mode)
{
    const int idx = blockIdx.x * 256 + threadIdx.x;
    if (idx >= M_TOTAL * D_MODEL) return;
    const int n = idx & (D_MODEL - 1);
    const int m = idx >> 10;
    const int s = m & (SEQ - 1);
    const int b = m >> 11;

    float val;
    if (mode == 2) {
        val = P[idx];
    } else {
        const int i = n & 511;
        const float2 p2 = *(const float2*)(P + (size_t)m * D_MODEL + 2 * i);
        const float freq = __expf(-(float)(2 * i) * (9.210340372f / 1024.0f));
        const float ang = (float)s * freq;
        float sn, c;
        __sincosf(ang, &sn, &c);
        const float xc = p2.x * c - p2.y * sn;
        val = (n < 512) ? xc : (xc * sn + p2.y * c);
        if (mode == 0) val *= 0.125f;
    }
    const int h = n >> 6, d = n & 63;
    out[(((size_t)(b * N_HEADS + h)) * SEQ + s) * HEAD_DIM + d] = (__bf16)val;
}

// ---------------------------------------------------------------------------
// Causal flash attention. 1 WG = 128 q-rows of one (b,h); 8 waves x 16 rows.
// K staged to LDS via async DMA; V staged transposed. Output bf16 [B,S,D].
// ---------------------------------------------------------------------------
#define QB 128
#define KB 32

__global__ __launch_bounds__(256)
void flash_attn(const __bf16* __restrict__ Qh,
                const __bf16* __restrict__ Kh,
                const __bf16* __restrict__ Vh,
                __bf16* __restrict__ O)
{
    __shared__ __bf16 Ks[KB * 72];
    __shared__ __bf16 Vt[64 * 40];
    __shared__ float  Sl[8][16 * 33];
    __shared__ __bf16 Pl[8][16 * 40];
    __shared__ float  Al[8][16];
    __shared__ float  Ll[8][16];

    const int tid  = threadIdx.x;
    const int lane = tid & 31;
    const int wave = tid >> 5;
    const int bh = blockIdx.y;
    const int b  = bh >> 4;
    const int h  = bh & 15;
    const int q0wg = blockIdx.x * QB;
    const int q0   = q0wg + wave * 16;

    const __bf16* Qp = Qh + (size_t)bh * SEQ * HEAD_DIM;
    const __bf16* Kp = Kh + (size_t)bh * SEQ * HEAD_DIM;
    const __bf16* Vp = Vh + (size_t)bh * SEQ * HEAD_DIM;

    FragBF qa[2];
    {
        const int mrow = lane & 15;
        const int k0   = (lane < 16) ? 0 : 8;
        const __bf16* qrow = Qp + (size_t)(q0 + mrow) * HEAD_DIM;
        qa[0].q[0] = *(const uint4*)(qrow + k0);
        qa[0].q[1] = *(const uint4*)(qrow + k0 + 16);
        qa[1].q[0] = *(const uint4*)(qrow + 32 + k0);
        qa[1].q[1] = *(const uint4*)(qrow + 32 + k0 + 16);
    }

    v8f oacc[4];
#pragma unroll
    for (int j = 0; j < 4; ++j)
#pragma unroll
        for (int r = 0; r < 8; ++r) oacc[j][r] = 0.0f;
    float m_i = -3.0e38f;
    float l_i = 0.0f;

    // staging map: thread -> one 8-elem chunk (key row = tid>>3, d col = (tid&7)*8)
    const int skr = tid >> 3;
    const int skc = (tid & 7) * 8;

    const int nkb = (q0wg + QB) / KB;
    for (int kb = 0; kb < nkb; ++kb) {
        const int kbase = kb * KB;
        __syncthreads();
        {
            // K tile: async DMA global->LDS (16B per lane)
            const unsigned voffK = (unsigned)(((kbase + skr) * HEAD_DIM + skc) * 2);
            async_copy_b128(lds_addr_of(&Ks[skr * 72 + skc]), Kp, voffK);
            // V tile transposed: manual
            const __bf16* vsrc = Vp + (size_t)(kbase + skr) * HEAD_DIM + skc;
#pragma unroll
            for (int j = 0; j < 8; ++j) Vt[(skc + j) * 40 + skr] = vsrc[j];
        }
        wait_async();
        __syncthreads();

        if (kbase <= q0 + 15) {   // wave-uniform causal skip
            const int nl = lane & 15;
            const int mb = (lane >> 4) * 8;
            // ---- S = Q K^T ----
            v8f sc[2];
#pragma unroll
            for (int t = 0; t < 2; ++t) {
#pragma unroll
                for (int r = 0; r < 8; ++r) sc[t][r] = 0.0f;
                const int khalf = (lane >> 4) * 8;
                FragBF kb0, kb1;
                const __bf16* krow = &Ks[(t * 16 + nl) * 72];
                kb0.q[0] = *(const uint4*)(krow + khalf);
                kb0.q[1] = *(const uint4*)(krow + khalf + 16);
                kb1.q[0] = *(const uint4*)(krow + 32 + khalf);
                kb1.q[1] = *(const uint4*)(krow + 32 + khalf + 16);
                sc[t] = __builtin_amdgcn_wmma_f32_16x16x32_bf16(
                    false, qa[0].v, false, kb0.v, (short)0, sc[t], false, false);
                sc[t] = __builtin_amdgcn_wmma_f32_16x16x32_bf16(
                    false, qa[1].v, false, kb1.v, (short)0, sc[t], false, false);
            }
            // ---- mask + spill ----
#pragma unroll
            for (int t = 0; t < 2; ++t)
#pragma unroll
                for (int r = 0; r < 8; ++r) {
                    const int mrw = mb + r;
                    const int kg = kbase + t * 16 + nl;
                    float v = sc[t][r];
                    if (kg > q0 + mrw) v = -3.0e38f;
                    Sl[wave][mrw * 33 + t * 16 + nl] = v;
                }
            // ---- online softmax ----
            {
                const int row = lane & 15;
                const int ch  = (lane >> 4) * 16;
                float sv[16];
                float lmax = -3.0e38f;
#pragma unroll
                for (int j = 0; j < 16; ++j) {
                    sv[j] = Sl[wave][row * 33 + ch + j];
                    lmax = fmaxf(lmax, sv[j]);
                }
                lmax = fmaxf(lmax, __shfl_xor(lmax, 16, 32));
                const float mnew  = fmaxf(m_i, lmax);
                const float alpha = __expf(m_i - mnew);
                float lsum = 0.0f;
#pragma unroll
                for (int j = 0; j < 16; ++j) {
                    const float p = __expf(sv[j] - mnew);
                    lsum += p;
                    Pl[wave][row * 40 + ch + j] = (__bf16)p;
                }
                lsum += __shfl_xor(lsum, 16, 32);
                l_i = alpha * l_i + lsum;
                m_i = mnew;
                if (lane < 16) Al[wave][row] = alpha;
            }
            // ---- rescale O, accumulate P * V ----
            {
                FragBF pa;
                const int k0 = (lane < 16) ? 0 : 8;
                const __bf16* prow = &Pl[wave][(lane & 15) * 40];
                pa.q[0] = *(const uint4*)(prow + k0);
                pa.q[1] = *(const uint4*)(prow + k0 + 16);
                float av[8];
#pragma unroll
                for (int r = 0; r < 8; ++r) av[r] = Al[wave][mb + r];
                const int k0b = (lane >> 4) * 8;
#pragma unroll
                for (int j = 0; j < 4; ++j) {
                    FragBF vb;
                    const __bf16* vrow = &Vt[(j * 16 + nl) * 40 + k0b];
                    vb.q[0] = *(const uint4*)(vrow);
                    vb.q[1] = *(const uint4*)(vrow + 16);
#pragma unroll
                    for (int r = 0; r < 8; ++r) oacc[j][r] *= av[r];
                    oacc[j] = __builtin_amdgcn_wmma_f32_16x16x32_bf16(
                        false, pa.v, false, vb.v, (short)0, oacc[j], false, false);
                }
            }
        }
    }
    // ---- finalize: O /= l, write bf16 [B,S,D] ----
    if (lane < 16) Ll[wave][lane] = 1.0f / l_i;
    const int nl = lane & 15;
    const int mb = (lane >> 4) * 8;
    float li[8];
#pragma unroll
    for (int r = 0; r < 8; ++r) li[r] = Ll[wave][mb + r];
#pragma unroll
    for (int j = 0; j < 4; ++j)
#pragma unroll
        for (int r = 0; r < 8; ++r) {
            const int mrw = q0 + mb + r;
            O[((size_t)b * SEQ + mrw) * D_MODEL + h * HEAD_DIM + j * 16 + nl] =
                (__bf16)(oacc[j][r] * li[r]);
        }
}

// ---------------------------------------------------------------------------
extern "C" void kernel_launch(void* const* d_in, const int* in_sizes, int n_in,
                              void* d_out, int out_size, void* d_ws, size_t ws_size,
                              hipStream_t stream)
{
    (void)in_sizes; (void)n_in; (void)out_size; (void)ws_size;
    const float* x  = (const float*)d_in[0];
    const float* wq = (const float*)d_in[1];
    const float* bq = (const float*)d_in[2];
    const float* wk = (const float*)d_in[3];
    const float* bk = (const float*)d_in[4];
    const float* wv = (const float*)d_in[5];
    const float* bv = (const float*)d_in[6];
    const float* wo = (const float*)d_in[7];
    const float* bo = (const float*)d_in[8];
    float* out = (float*)d_out;

    char* ws = (char*)d_ws;
    size_t off = 0;
    const size_t NE = (size_t)M_TOTAL * D_MODEL;   // 4M
    const size_t NW = (size_t)D_MODEL * D_MODEL;   // 1M
    float*  Pq  = (float*)(ws + off);  off += NE * 4;
    float*  Pk  = (float*)(ws + off);  off += NE * 4;
    float*  Pv  = (float*)(ws + off);  off += NE * 4;
    __bf16* Qh  = (__bf16*)(ws + off); off += NE * 2;
    __bf16* Kh  = (__bf16*)(ws + off); off += NE * 2;
    __bf16* Vh  = (__bf16*)(ws + off); off += NE * 2;
    __bf16* xb  = (__bf16*)(ws + off); off += NE * 2;
    __bf16* wqb = (__bf16*)(ws + off); off += NW * 2;
    __bf16* wkb = (__bf16*)(ws + off); off += NW * 2;
    __bf16* wvb = (__bf16*)(ws + off); off += NW * 2;
    __bf16* wob = (__bf16*)(ws + off); off += NW * 2;
    __bf16* Ob  = (__bf16*)(ws + off); off += NE * 2;

    // preconvert activations + weights to bf16
    to_bf16<<<(int)(NE / 4 + 255) / 256, 256, 0, stream>>>(x, xb, (int)NE);
    to_bf16<<<(int)(NW / 4 + 255) / 256, 256, 0, stream>>>(wq, wqb, (int)NW);
    to_bf16<<<(int)(NW / 4 + 255) / 256, 256, 0, stream>>>(wk, wkb, (int)NW);
    to_bf16<<<(int)(NW / 4 + 255) / 256, 256, 0, stream>>>(wv, wvb, (int)NW);
    to_bf16<<<(int)(NW / 4 + 255) / 256, 256, 0, stream>>>(wo, wob, (int)NW);

    dim3 gg(D_MODEL / TB_N, M_TOTAL / TB_M);   // (8, 32)
    gemm_nt_bias<<<gg, 256, 0, stream>>>(xb, wqb, bq, Pq, M_TOTAL, D_MODEL, D_MODEL);
    gemm_nt_bias<<<gg, 256, 0, stream>>>(xb, wkb, bk, Pk, M_TOTAL, D_MODEL, D_MODEL);
    gemm_nt_bias<<<gg, 256, 0, stream>>>(xb, wvb, bv, Pv, M_TOTAL, D_MODEL, D_MODEL);

    const int total = M_TOTAL * D_MODEL;
    rope_pack<<<(total + 255) / 256, 256, 0, stream>>>(Pq, Qh, 0);
    rope_pack<<<(total + 255) / 256, 256, 0, stream>>>(Pk, Kh, 1);
    rope_pack<<<(total + 255) / 256, 256, 0, stream>>>(Pv, Vh, 2);

    flash_attn<<<dim3(SEQ / QB, BATCH * N_HEADS), 256, 0, stream>>>(Qh, Kh, Vh, Ob);

    gemm_nt_bias<<<gg, 256, 0, stream>>>(Ob, wob, bo, out, M_TOTAL, D_MODEL, D_MODEL);
}
